// MultiHeadAttention_32658931319494
// MI455X (gfx1250) — matrix operand order repaired and checked by reading, not compiled, not run on hardware
//
#include <hip/hip_runtime.h>

typedef __bf16 bf16;
typedef __attribute__((ext_vector_type(16))) __bf16 v16bf;
typedef __attribute__((ext_vector_type(8)))  __bf16 v8bf;
typedef __attribute__((ext_vector_type(8)))  float  v8f;
typedef __attribute__((ext_vector_type(4)))  int    v4i;

// address-space qualified 16-byte vector types for the async-LDS builtin
typedef __attribute__((address_space(1))) v4i gv4i;  // global (__device__)
typedef __attribute__((address_space(3))) v4i lv4i;  // LDS (__shared__)

#define WMMA_BF16(a, b, c) \
  __builtin_amdgcn_wmma_f32_16x16x32_bf16(false, (a), false, (b), (short)0, (c), false, false)

// ---------------------------------------------------------------------------
// async global->LDS copy (ASYNCcnt path) with safe fallback
// ---------------------------------------------------------------------------
#if __has_builtin(__builtin_amdgcn_global_load_async_to_lds_b128)
#define HAS_ASYNC_LDS 1
#else
#define HAS_ASYNC_LDS 0
#endif

__device__ __forceinline__ void copy16_g2l(const bf16* g, bf16* l) {
#if HAS_ASYNC_LDS
  __builtin_amdgcn_global_load_async_to_lds_b128(
      (gv4i*)(void*)g, (lv4i*)(void*)l, 0, 0);
#else
  *(uint4*)l = *(const uint4*)g;
#endif
}

__device__ __forceinline__ void wait_async() {
#if HAS_ASYNC_LDS
#if __has_builtin(__builtin_amdgcn_s_wait_asynccnt)
  __builtin_amdgcn_s_wait_asynccnt(0);
#else
  asm volatile("s_wait_asynccnt 0x0" ::: "memory");
#endif
#endif
}

// ---------------------------------------------------------------------------
// lane-xor reductions over the 16 lanes of a half-wave (rows of WMMA C-tile)
// ds_swizzle_b32 group-of-32: offset = xor<<10 | 0x1f (no address VGPR needed)
// ---------------------------------------------------------------------------
#if __has_builtin(__builtin_amdgcn_ds_swizzle)
#define SWZ_XOR(x, imm) \
  __int_as_float(__builtin_amdgcn_ds_swizzle(__float_as_int(x), ((imm) << 10) | 0x1f))
#else
#define SWZ_XOR(x, imm) __shfl_xor((x), (imm), 32)
#endif

__device__ __forceinline__ float red_max16(float x) {
  x = fmaxf(x, SWZ_XOR(x, 1));
  x = fmaxf(x, SWZ_XOR(x, 2));
  x = fmaxf(x, SWZ_XOR(x, 4));
  x = fmaxf(x, SWZ_XOR(x, 8));
  return x;
}
__device__ __forceinline__ float red_sum16(float x) {
  x += SWZ_XOR(x, 1);
  x += SWZ_XOR(x, 2);
  x += SWZ_XOR(x, 4);
  x += SWZ_XOR(x, 8);
  return x;
}

// ---------------------------------------------------------------------------
// f32 -> bf16 conversion (streaming)
// ---------------------------------------------------------------------------
__global__ void cvt_f32_bf16(const float* __restrict__ src, bf16* __restrict__ dst, long n) {
  long i = (long)blockIdx.x * blockDim.x + threadIdx.x;
  long stride = (long)gridDim.x * blockDim.x;
  for (; i < n; i += stride) dst[i] = (bf16)src[i];
}

// ---------------------------------------------------------------------------
// bf16 WMMA GEMM: C[M,N] = A[M,K] * B[K,N] + bias[N]
// Block tile 256x64, 8 waves (4x2), wave tile 64x32 (4x2 frags), K step 64.
// Double-buffered pipeline: async A(t+1)->LDS + global B(t+1)->regs issued
// before computing tile t; B regs scattered (transposed) after the WMMAs.
// ---------------------------------------------------------------------------
template <bool OUT_F32>
__global__ __launch_bounds__(256) void gemm_bf16(const bf16* __restrict__ A,
                                                 const bf16* __restrict__ B,
                                                 const float* __restrict__ bias,
                                                 void* __restrict__ Cout,
                                                 int M, int N, int K) {
  __shared__ alignas(32) bf16 ldsA[2][256 * 64];  // [m][k]  2 x 32 KB
  __shared__ alignas(32) bf16 ldsBt[2][64 * 64];  // [n][k]  2 x  8 KB

  const int tid = threadIdx.x;
  const int lane = tid & 31;
  const int wid = tid >> 5;
  const int wm = wid >> 1;  // 0..3
  const int wn = wid & 1;   // 0..1
  const int h  = lane >> 4; // lane half
  const int ln = lane & 15;
  const int bm = blockIdx.y * 256;
  const int bn = blockIdx.x * 64;

  // per-thread load coordinates
  const int arow[8] = {(tid * 8 + 0) >> 3, (tid * 8 + 1) >> 3, (tid * 8 + 2) >> 3,
                       (tid * 8 + 3) >> 3, (tid * 8 + 4) >> 3, (tid * 8 + 5) >> 3,
                       (tid * 8 + 6) >> 3, (tid * 8 + 7) >> 3};
  v8f acc[4][2] = {};
  const int T = K / 64;

  // ---- prologue: stage tile 0 ----
#pragma unroll
  for (int i = 0; i < 8; ++i) {
    int idx = tid * 8 + i;
    int row = idx >> 3, c = (idx & 7) * 8;
    copy16_g2l(&A[(long)(bm + row) * K + c], &ldsA[0][row * 64 + c]);
  }
  {
    uint4 breg[2];
#pragma unroll
    for (int i = 0; i < 2; ++i) {
      int idx = tid * 2 + i;
      int row = idx >> 3, c = (idx & 7) * 8;
      breg[i] = *(const uint4*)&B[(long)row * N + bn + c];
    }
#pragma unroll
    for (int i = 0; i < 2; ++i) {
      int idx = tid * 2 + i;
      int row = idx >> 3, c = (idx & 7) * 8;
      const bf16* p = (const bf16*)&breg[i];
#pragma unroll
      for (int j = 0; j < 8; ++j) ldsBt[0][(c + j) * 64 + row] = p[j];
    }
  }
  wait_async();
  __syncthreads();

  for (int t = 0; t < T; ++t) {
    const int cur = t & 1, nxt = cur ^ 1;
    const bool more = (t + 1) < T;
    const int k1 = (t + 1) * 64;

    uint4 breg[2];
    if (more) {
      // issue next tile's copies before computing the current one
#pragma unroll
      for (int i = 0; i < 8; ++i) {
        int idx = tid * 8 + i;
        int row = idx >> 3, c = (idx & 7) * 8;
        copy16_g2l(&A[(long)(bm + row) * K + k1 + c], &ldsA[nxt][row * 64 + c]);
      }
#pragma unroll
      for (int i = 0; i < 2; ++i) {
        int idx = tid * 2 + i;
        int row = idx >> 3, c = (idx & 7) * 8;
        breg[i] = *(const uint4*)&B[(long)(k1 + row) * N + bn + c];
      }
    }

    // ---- compute current tile ----
#pragma unroll
    for (int kk = 0; kk < 64; kk += 32) {
      v16bf bfrag[2];
#pragma unroll
      for (int fn = 0; fn < 2; ++fn) {
        int c = wn * 32 + fn * 16 + ln;  // B-layout: N = lane&15, K = 16h + e
        bfrag[fn] = *(const v16bf*)&ldsBt[cur][c * 64 + kk + 16 * h];
      }
#pragma unroll
      for (int fm = 0; fm < 4; ++fm) {
        int r = wm * 64 + fm * 16 + ln;  // A-layout: M = lane&15
        v8bf lo = *(const v8bf*)&ldsA[cur][r * 64 + kk + 16 * h];      // K=16h+j
        v8bf hi = *(const v8bf*)&ldsA[cur][r * 64 + kk + 16 * h + 8];  // K=16h+8+j
        v16bf afrag;
#pragma unroll
        for (int j = 0; j < 8; ++j) { afrag[j] = lo[j]; afrag[8 + j] = hi[j]; }
#pragma unroll
        for (int fn = 0; fn < 2; ++fn)
          acc[fm][fn] = WMMA_BF16(afrag, bfrag[fn], acc[fm][fn]);
      }
    }

    if (more) {
      // scatter next B tile (transposed) into the idle buffer
#pragma unroll
      for (int i = 0; i < 2; ++i) {
        int idx = tid * 2 + i;
        int row = idx >> 3, c = (idx & 7) * 8;
        const bf16* p = (const bf16*)&breg[i];
#pragma unroll
        for (int j = 0; j < 8; ++j) ldsBt[nxt][(c + j) * 64 + row] = p[j];
      }
    }
    wait_async();
    __syncthreads();
  }
  (void)arow;

#pragma unroll
  for (int fm = 0; fm < 4; ++fm)
#pragma unroll
    for (int fn = 0; fn < 2; ++fn) {
      int col = bn + wn * 32 + fn * 16 + ln;
      float bv = bias[col];
#pragma unroll
      for (int r = 0; r < 8; ++r) {  // C-layout: M = r + 8h, N = lane&15
        int row = bm + wm * 64 + fm * 16 + r + 8 * h;
        float v = acc[fm][fn][r] + bv;
        if (OUT_F32)
          ((float*)Cout)[(long)row * N + col] = v;
        else
          ((bf16*)Cout)[(long)row * N + col] = (bf16)v;
      }
    }
}

// ---------------------------------------------------------------------------
// Flash attention chunk body, specialized on whether causal masking applies.
// ---------------------------------------------------------------------------
template <bool MASKED>
__device__ __forceinline__ void attn_chunk(const v16bf (&qf)[4], v8f (&o)[8],
                                           float (&mrow)[8], float (&lrow)[8],
                                           const bf16* __restrict__ ldsK,
                                           const bf16* __restrict__ ldsVt,
                                           bf16* __restrict__ ldsPw,
                                           int kbase, int qbase, int h, int ln) {
  const float scale = 0.08838834764831845f;  // 1/sqrt(128)
  // scores: S[16 x 32] over 4 d-chunks; s0 = keys 0..15, s1 = keys 16..31
  v8f s0 = {}, s1 = {};
#pragma unroll
  for (int c = 0; c < 4; ++c) {
    v16bf b0 = *(const v16bf*)&ldsK[ln * 128 + c * 32 + 16 * h];
    v16bf b1 = *(const v16bf*)&ldsK[(16 + ln) * 128 + c * 32 + 16 * h];
    s0 = WMMA_BF16(qf[c], b0, s0);
    s1 = WMMA_BF16(qf[c], b1, s1);
  }

  float alpha[8];
#pragma unroll
  for (int r = 0; r < 8; ++r) {
    float v0 = s0[r] * scale;
    float v1 = s1[r] * scale;
    if (MASKED) {
      int qrow = qbase + r + 8 * h;
      if (kbase + ln > qrow)      v0 = -1e30f;
      if (kbase + 16 + ln > qrow) v1 = -1e30f;
    }
    float rm = red_max16(fmaxf(v0, v1));
    float nm = fmaxf(mrow[r], rm);
    alpha[r] = __expf(mrow[r] - nm);
    mrow[r] = nm;
    float p0 = __expf(v0 - nm);
    float p1 = __expf(v1 - nm);
    lrow[r] = lrow[r] * alpha[r] + red_sum16(p0 + p1);
    // stash P (C-layout -> LDS) for re-distribution into A-frag layout
    ldsPw[(r + 8 * h) * 32 + ln]      = (bf16)p0;
    ldsPw[(r + 8 * h) * 32 + 16 + ln] = (bf16)p1;
  }
#pragma unroll
  for (int cc = 0; cc < 8; ++cc)
#pragma unroll
    for (int r = 0; r < 8; ++r) o[cc][r] *= alpha[r];

  // P fragment (A-layout, 16x32)
  v16bf pf;
  {
    v8bf lo = *(const v8bf*)&ldsPw[ln * 32 + 16 * h];
    v8bf hi = *(const v8bf*)&ldsPw[ln * 32 + 16 * h + 8];
#pragma unroll
    for (int j = 0; j < 8; ++j) { pf[j] = lo[j]; pf[8 + j] = hi[j]; }
  }
  // O += P * V, 8 output d-chunks of 16
#pragma unroll
  for (int cc = 0; cc < 8; ++cc) {
    v16bf vf = *(const v16bf*)&ldsVt[(cc * 16 + ln) * 32 + 16 * h];
    o[cc] = WMMA_BF16(pf, vf, o[cc]);
  }
}

// ---------------------------------------------------------------------------
// Flash attention (causal). 4 waves per block; each wave owns a 16-row Q tile,
// the block covers 64 query rows and shares K/V LDS tiles across waves.
// Double-buffered: async K(t+1) + V(t+1)->regs issued before computing t.
// QKV packed bf16 [B*S, 6144]: Q at +0, K at +2048, V at +4096 (per row).
// ---------------------------------------------------------------------------
__global__ __launch_bounds__(128) void attn_kernel(const bf16* __restrict__ QKV,
                                                   bf16* __restrict__ O) {
  const int qb = blockIdx.x;    // 0..31 (64-row q block)
  const int bh = blockIdx.y;    // 0..63
  const int b = bh >> 4, head = bh & 15;
  const int tid = threadIdx.x;
  const int lane = tid & 31;
  const int wave = tid >> 5;    // 0..3
  const int h = lane >> 4, ln = lane & 15;
  const int qbase = qb * 64 + wave * 16;  // this wave's first query row

  const long baseQ = (long)b * 2048 * 6144 + (long)head * 128;
  const long baseK = baseQ + 2048;
  const long baseV = baseQ + 4096;

  __shared__ alignas(32) bf16 ldsQ[4][16 * 128];   // per-wave [q][d]  16 KB
  __shared__ alignas(32) bf16 ldsK[2][32 * 128];   // [key][d]  2 x 8 KB
  __shared__ alignas(32) bf16 ldsVt[2][128 * 32];  // [d][key]  2 x 8 KB
  __shared__ alignas(32) bf16 ldsP[4][16 * 32];    // per-wave [q][key] 4 KB

  // load coordinates for cooperative K/V staging (4 x uint4 per thread)
  const int vrow = tid >> 4;          // 0..7  (+i*8)
  const int vcol = (tid & 15) * 8;    // d col

  // Q tile 16x128 -> LDS (each wave async-copies its own tile)
#pragma unroll
  for (int i = 0; i < 8; ++i) {
    int idx = i * 32 + lane;   // 0..255
    int row = idx >> 4;        // 0..15
    int c = (idx & 15) * 8;
    copy16_g2l(&QKV[baseQ + (long)(qbase + row) * 6144 + c],
               &ldsQ[wave][row * 128 + c]);
  }
  // ---- prologue: stage K/V chunk 0 ----
#pragma unroll
  for (int i = 0; i < 4; ++i) {
    int row = vrow + i * 8;
    copy16_g2l(&QKV[baseK + (long)row * 6144 + vcol], &ldsK[0][row * 128 + vcol]);
  }
  {
    uint4 vreg[4];
#pragma unroll
    for (int i = 0; i < 4; ++i)
      vreg[i] = *(const uint4*)&QKV[baseV + (long)(vrow + i * 8) * 6144 + vcol];
#pragma unroll
    for (int i = 0; i < 4; ++i) {
      const bf16* p = (const bf16*)&vreg[i];
#pragma unroll
      for (int j = 0; j < 8; ++j) ldsVt[0][(vcol + j) * 32 + vrow + i * 8] = p[j];
    }
  }
  wait_async();

  // Q fragments: 4 d-chunks of 32
  v16bf qf[4];
#pragma unroll
  for (int c = 0; c < 4; ++c) {
    v8bf lo = *(const v8bf*)&ldsQ[wave][ln * 128 + c * 32 + 16 * h];
    v8bf hi = *(const v8bf*)&ldsQ[wave][ln * 128 + c * 32 + 16 * h + 8];
#pragma unroll
    for (int j = 0; j < 8; ++j) { qf[c][j] = lo[j]; qf[c][8 + j] = hi[j]; }
  }

  v8f o[8] = {};
  float mrow[8], lrow[8];
#pragma unroll
  for (int r = 0; r < 8; ++r) { mrow[r] = -1e30f; lrow[r] = 0.0f; }

  const int nch = qb * 2 + 2;  // key chunks needed by the block
  __syncthreads();

  for (int kc = 0; kc < nch; ++kc) {
    const int cur = kc & 1, nxt = cur ^ 1;
    const bool more = (kc + 1) < nch;
    const int kbase = kc * 32;
    const int kbase1 = kbase + 32;

    uint4 vreg[4];
    if (more) {
      // issue next chunk's K async copy + V global loads before computing
#pragma unroll
      for (int i = 0; i < 4; ++i) {
        int row = vrow + i * 8;
        copy16_g2l(&QKV[baseK + (long)(kbase1 + row) * 6144 + vcol],
                   &ldsK[nxt][row * 128 + vcol]);
      }
#pragma unroll
      for (int i = 0; i < 4; ++i)
        vreg[i] = *(const uint4*)&QKV[baseV + (long)(kbase1 + vrow + i * 8) * 6144 + vcol];
    }

    // ---- compute current chunk ----
    if (kbase + 31 <= qbase) {
      attn_chunk<false>(qf, o, mrow, lrow, ldsK[cur], ldsVt[cur], ldsP[wave],
                        kbase, qbase, h, ln);
    } else if (kbase <= qbase + 15) {
      attn_chunk<true>(qf, o, mrow, lrow, ldsK[cur], ldsVt[cur], ldsP[wave],
                       kbase, qbase, h, ln);
    }

    if (more) {
      // scatter next V tile (transposed) into the idle buffer
#pragma unroll
      for (int i = 0; i < 4; ++i) {
        const bf16* p = (const bf16*)&vreg[i];
#pragma unroll
        for (int j = 0; j < 8; ++j) ldsVt[nxt][(vcol + j) * 32 + vrow + i * 8] = p[j];
      }
    }
    wait_async();
    __syncthreads();
  }

#pragma unroll
  for (int r = 0; r < 8; ++r) lrow[r] = 1.0f / lrow[r];
#pragma unroll
  for (int cc = 0; cc < 8; ++cc)
#pragma unroll
    for (int r = 0; r < 8; ++r) {
      long row = (long)b * 2048 + qbase + r + 8 * h;
      O[row * 2048 + (long)head * 128 + cc * 16 + ln] = (bf16)(o[cc][r] * lrow[r]);
    }
}

// ---------------------------------------------------------------------------
// launcher
// ---------------------------------------------------------------------------
extern "C" void kernel_launch(void* const* d_in, const int* in_sizes, int n_in,
                              void* d_out, int out_size, void* d_ws, size_t ws_size,
                              hipStream_t stream) {
  const float* x     = (const float*)d_in[0];  // [4,2048,2048]
  const float* W_qkv = (const float*)d_in[1];  // [2048,6144]
  const float* b_qkv = (const float*)d_in[2];  // [6144]
  const float* W_out = (const float*)d_in[3];  // [2048,2048]
  const float* b_out = (const float*)d_in[4];  // [2048]
  float* out = (float*)d_out;                  // [4,2048,2048] fp32

  const long MS = 4L * 2048;  // 8192 rows
  char* ws = (char*)d_ws;
  bf16* Xb   = (bf16*)ws;  ws += (size_t)MS * 2048 * 2;   // 32 MB
  bf16* Wqb  = (bf16*)ws;  ws += (size_t)2048 * 6144 * 2; // 24 MB
  bf16* Wob  = (bf16*)ws;  ws += (size_t)2048 * 2048 * 2; //  8 MB
  bf16* QKV  = (bf16*)ws;  ws += (size_t)MS * 6144 * 2;   // 96 MB
  bf16* Obuf = (bf16*)ws;                                 // 32 MB

  cvt_f32_bf16<<<2048, 256, 0, stream>>>(x, Xb, MS * 2048);
  cvt_f32_bf16<<<2048, 256, 0, stream>>>(W_qkv, Wqb, 2048L * 6144);
  cvt_f32_bf16<<<2048, 256, 0, stream>>>(W_out, Wob, 2048L * 2048);

  // QKV = Xb * Wqb + b_qkv   (bf16 out)
  gemm_bf16<false><<<dim3(6144 / 64, MS / 256), 256, 0, stream>>>(
      Xb, Wqb, b_qkv, QKV, (int)MS, 6144, 2048);

  // flash attention -> Obuf bf16 [8192, 2048]
  attn_kernel<<<dim3(32, 64), 128, 0, stream>>>(QKV, Obuf);

  // out = Obuf * Wob + b_out  (fp32 out)
  gemm_bf16<true><<<dim3(2048 / 64, MS / 256), 256, 0, stream>>>(
      Obuf, Wob, b_out, out, (int)MS, 2048, 2048);
}
